// GeoNet_74371653698224
// MI455X (gfx1250) — matrix operand order repaired and checked by reading, not compiled, not run on hardware
//
#include <hip/hip_runtime.h>
#include <math.h>

typedef __attribute__((ext_vector_type(16))) _Float16 v16h;
typedef __attribute__((ext_vector_type(8)))  float    v8f;

#define H_FULL 256
#define W_FULL 832
#define OUT_C  21
#define PI_F   3.14159265358979f

#if defined(__HIP_DEVICE_COMPILE__) && defined(__gfx1250__) && \
    __has_builtin(__builtin_amdgcn_global_load_async_to_lds_b32) && \
    __has_builtin(__builtin_amdgcn_s_wait_asynccnt)
#define USE_ASYNC_LDS 1
#else
#define USE_ASYNC_LDS 0
#endif

// ---- workspace layout (float offsets) ----
#define ACC_OFF  0          // 32 accumulators (+aux slot 20)
#define PF_OFF   64         // 4*2*4 * 16
#define PB_OFF   576
#define KI_OFF   1088       // 4*4 * 9
#define BIG_OFF  1280
#define IMG3_SZ  (8*H_FULL*W_FULL*3)
#define FLOW_SZ  (8*H_FULL*W_FULL*2)
#define TGT_OFF  (BIG_OFF)
#define SRC_OFF  (TGT_OFF + IMG3_SZ)
#define FR_OFF   (SRC_OFF + IMG3_SZ)
#define BR_OFF   (FR_OFF + FLOW_SZ)
#define FF_OFF   (BR_OFF + FLOW_SZ)
#define BF_OFF   (FF_OFF + FLOW_SZ)

// -------------------- helpers --------------------
struct Taps { int ix0, ix1, iy0, iy1; float wx0, wx1, wy0, wy1; };

__device__ inline Taps mkTaps(float xx, float yy, int Wd, int Hd) {
  Taps t;
  float fx0 = floorf(xx), fy0 = floorf(yy);
  float x0s = fminf(fmaxf(fx0,      0.f), (float)(Wd-1));
  float x1s = fminf(fmaxf(fx0+1.f,  0.f), (float)(Wd-1));
  float y0s = fminf(fmaxf(fy0,      0.f), (float)(Hd-1));
  float y1s = fminf(fmaxf(fy0+1.f,  0.f), (float)(Hd-1));
  t.wx0 = x1s - xx; t.wx1 = xx - x0s;
  t.wy0 = y1s - yy; t.wy1 = yy - y0s;
  t.ix0 = (int)x0s; t.ix1 = (int)x1s; t.iy0 = (int)y0s; t.iy1 = (int)y1s;
  return t;
}

__device__ inline float gatherC(const float* __restrict__ p, int b, int c, int C,
                                int Hd, int Wd, const Taps& t) {
  const float* base = p + ((size_t)b * Hd) * Wd * C + c;
  float g00 = base[((size_t)t.iy0 * Wd + t.ix0) * C];
  float g01 = base[((size_t)t.iy0 * Wd + t.ix1) * C];
  float g10 = base[((size_t)t.iy1 * Wd + t.ix0) * C];
  float g11 = base[((size_t)t.iy1 * Wd + t.ix1) * C];
  return t.wy0 * (t.wx0 * g00 + t.wx1 * g01) + t.wy1 * (t.wx0 * g10 + t.wx1 * g11);
}

__device__ inline float l2n2(float a, float b) {
  float aa = fabsf(a) + 1e-10f, bb = fabsf(b) + 1e-10f;
  return sqrtf(aa * aa + bb * bb);
}

__device__ inline float waveSum(float v) {
  for (int o = 16; o > 0; o >>= 1) v += __shfl_down(v, o, 32);
  return v;
}

__device__ inline void mm3(const float* A, const float* B, float* C) {
  for (int r = 0; r < 3; ++r)
    for (int c = 0; c < 3; ++c) {
      float a = 0.f;
      for (int k = 0; k < 3; ++k) a += A[r*3+k] * B[k*3+c];
      C[r*3+c] = a;
    }
}

// -------------------- kernels --------------------
__global__ void k_zero(float* __restrict__ acc) {
  if (threadIdx.x < 32) acc[threadIdx.x] = 0.f;
}

// one thread per (b, src, scale): P_fwd = K4*T, P_bwd = K4*inv(T), plus Kinv
__global__ void k_pose(const float* __restrict__ intr, const float* __restrict__ poses,
                       float* __restrict__ Pf, float* __restrict__ Pb,
                       float* __restrict__ Kinv) {
  int t = threadIdx.x;
  if (t >= 32) return;
  int s = t & 3, i = (t >> 2) & 1, b = t >> 3;
  const float* v = poses + (b * 2 + i) * 6;
  float rx = fminf(fmaxf(v[3], -PI_F), PI_F);
  float ry = fminf(fmaxf(v[4], -PI_F), PI_F);
  float rz = fminf(fmaxf(v[5], -PI_F), PI_F);
  float cz = cosf(rz), sz = sinf(rz);
  float cy = cosf(ry), sy = sinf(ry);
  float cx = cosf(rx), sx = sinf(rx);
  float Rz[9] = {cz,-sz,0.f,  sz,cz,0.f,  0.f,0.f,1.f};
  float Ry[9] = {cy,0.f,sy,   0.f,1.f,0.f, -sy,0.f,cy};
  float Rx[9] = {1.f,0.f,0.f, 0.f,cx,-sx, 0.f,sx,cx};
  float M[9], R[9];
  mm3(Rz, Ry, M); mm3(M, Rx, R);
  float tr[3] = {v[0], v[1], v[2]};
  const float* K = intr + ((size_t)b * 4 + s) * 9;

  float* pf = Pf + (((size_t)b * 2 + i) * 4 + s) * 16;
  for (int r = 0; r < 3; ++r) {
    for (int c = 0; c < 3; ++c) {
      float a = 0.f;
      for (int k = 0; k < 3; ++k) a += K[r*3+k] * R[k*3+c];
      pf[r*4+c] = a;
    }
    float a = 0.f;
    for (int k = 0; k < 3; ++k) a += K[r*3+k] * tr[k];
    pf[r*4+3] = a;
  }
  pf[12] = 0.f; pf[13] = 0.f; pf[14] = 0.f; pf[15] = 1.f;

  // inv(T) = [R^T | -R^T t]
  float Rt[9], ti[3];
  for (int r = 0; r < 3; ++r)
    for (int c = 0; c < 3; ++c) Rt[r*3+c] = R[c*3+r];
  for (int r = 0; r < 3; ++r)
    ti[r] = -(Rt[r*3+0]*tr[0] + Rt[r*3+1]*tr[1] + Rt[r*3+2]*tr[2]);

  float* pb = Pb + (((size_t)b * 2 + i) * 4 + s) * 16;
  for (int r = 0; r < 3; ++r) {
    for (int c = 0; c < 3; ++c) {
      float a = 0.f;
      for (int k = 0; k < 3; ++k) a += K[r*3+k] * Rt[k*3+c];
      pb[r*4+c] = a;
    }
    float a = 0.f;
    for (int k = 0; k < 3; ++k) a += K[r*3+k] * ti[k];
    pb[r*4+3] = a;
  }
  pb[12] = 0.f; pb[13] = 0.f; pb[14] = 0.f; pb[15] = 1.f;

  if (i == 0) {
    float* ki = Kinv + ((size_t)b * 4 + s) * 9;
    float a00=K[0],a01=K[1],a02=K[2],a10=K[3],a11=K[4],a12=K[5],a20=K[6],a21=K[7],a22=K[8];
    float det = a00*(a11*a22-a12*a21) - a01*(a10*a22-a12*a20) + a02*(a10*a21-a11*a20);
    float id = 1.f / det;
    ki[0]=(a11*a22-a12*a21)*id; ki[1]=(a02*a21-a01*a22)*id; ki[2]=(a01*a12-a02*a11)*id;
    ki[3]=(a12*a20-a10*a22)*id; ki[4]=(a00*a22-a02*a20)*id; ki[5]=(a02*a10-a00*a12)*id;
    ki[6]=(a10*a21-a11*a20)*id; ki[7]=(a01*a20-a00*a21)*id; ki[8]=(a00*a11-a01*a10)*id;
  }
}

// fused: pyramid resizes (+2x-1 normalize), depth, rigid & full flows per scale
__global__ void k_prep(const float* __restrict__ tgt, const float* __restrict__ src,
                       const float* __restrict__ disp, const float* __restrict__ fres,
                       const float* __restrict__ bres,
                       const float* __restrict__ Pf, const float* __restrict__ Pb,
                       const float* __restrict__ Kinv,
                       float* __restrict__ tgtA, float* __restrict__ srcA,
                       float* __restrict__ FR, float* __restrict__ BR,
                       float* __restrict__ FF, float* __restrict__ BF,
                       int s, int Hs, int Ws) {
  int idx = blockIdx.x * blockDim.x + threadIdx.x;
  if (idx >= Hs * Ws) return;
  int bb = blockIdx.y;
  int b = bb & 3, i = bb >> 2;
  int gr = idx / Ws, gc = idx % Ws;
  float f = (float)(1 << s);
  float sy = ((float)gr + 0.5f) * f - 0.5f;
  float sx = ((float)gc + 0.5f) * f - 0.5f;
  Taps t = mkTaps(sx, sy, W_FULL, H_FULL);
  size_t po = ((size_t)bb * Hs + gr) * Ws + gc;

  // streaming read-ahead on the widest input
  __builtin_prefetch(src + (((size_t)b * H_FULL + t.iy0) * W_FULL + t.ix0) * 6, 0, 1);

  for (int c = 0; c < 3; ++c) {
    tgtA[po*3 + c] = 2.f * gatherC(tgt, b, c,       3, H_FULL, W_FULL, t) - 1.f;
    srcA[po*3 + c] = 2.f * gatherC(src, b, 3*i + c, 6, H_FULL, W_FULL, t) - 1.f;
  }
  float df = 1.f / gatherC(disp, b,           0, 1, H_FULL, W_FULL, t);
  float db = 1.f / gatherC(disp, 4*(i+1) + b, 0, 1, H_FULL, W_FULL, t);

  const float* KI = Kinv + ((size_t)b * 4 + s) * 9;
  const float* PF = Pf + (((size_t)b * 2 + i) * 4 + s) * 16;
  const float* PB = Pb + (((size_t)b * 2 + i) * 4 + s) * 16;
  float u = (float)gc, v = (float)gr;
  float cxr = KI[0]*u + KI[1]*v + KI[2];
  float cyr = KI[3]*u + KI[4]*v + KI[5];
  float czr = KI[6]*u + KI[7]*v + KI[8];

  float X0 = df*cxr, Y0 = df*cyr, Z0 = df*czr;
  float ux = PF[0]*X0 + PF[1]*Y0 + PF[2]*Z0  + PF[3];
  float uy = PF[4]*X0 + PF[5]*Y0 + PF[6]*Z0  + PF[7];
  float uz = PF[8]*X0 + PF[9]*Y0 + PF[10]*Z0 + PF[11];
  float frx = ux / (uz + 1e-10f) - u;
  float fry = uy / (uz + 1e-10f) - v;

  float X1 = db*cxr, Y1 = db*cyr, Z1 = db*czr;
  float vx = PB[0]*X1 + PB[1]*Y1 + PB[2]*Z1  + PB[3];
  float vy = PB[4]*X1 + PB[5]*Y1 + PB[6]*Z1  + PB[7];
  float vz = PB[8]*X1 + PB[9]*Y1 + PB[10]*Z1 + PB[11];
  float brx = vx / (vz + 1e-10f) - u;
  float bry = vy / (vz + 1e-10f) - v;

  FR[po*2] = frx; FR[po*2+1] = fry;
  BR[po*2] = brx; BR[po*2+1] = bry;

  float rfx = gatherC(fres, bb, 0, 2, H_FULL, W_FULL, t) / f;
  float rfy = gatherC(fres, bb, 1, 2, H_FULL, W_FULL, t) / f;
  float rbx = gatherC(bres, bb, 0, 2, H_FULL, W_FULL, t) / f;
  float rby = gatherC(bres, bb, 1, 2, H_FULL, W_FULL, t) / f;
  FF[po*2] = frx + rfx; FF[po*2+1] = fry + rfy;
  BF[po*2] = brx + rbx; BF[po*2+1] = bry + rby;
}

// One wave per 16x16 tile, one (batch, channel) per blockIdx.y.
// Warp-on-the-fly into LDS (with halo), horizontal 3-tap sums in VALU,
// vertical 3-tap pass as V_WMMA_F32_16X16X32_F16 with a constant banded A.
// The reference-image tile is copied via GLOBAL_LOAD_ASYNC_TO_LDS (ASYNCcnt)
// when the toolchain exposes it, overlapping with the VALU warp gather.
__global__ __launch_bounds__(32) void k_ssim(
    const float* __restrict__ img, const float* __restrict__ ref,
    const float* __restrict__ flow, float* __restrict__ out,
    float* __restrict__ acc, int accIdx, int Hs, int Ws,
    int chbase, int isS0, int tilesX) {
  __shared__ float X[32][20];
  __shared__ float Y[32][20];
  const int lane = threadIdx.x;
  const int tile = blockIdx.x;
  const int ty = tile / tilesX, tx = tile % tilesX;
  const int bc = blockIdx.y;
  const int bb = bc / 3, c = bc % 3;
  const int r0 = ty * 16, c0 = tx * 16;

  // rows r0-1 .. r0+30, cols c0-1 .. c0+16 ; zero-pad outside (SAME pooling)
  for (int idx = lane; idx < 32 * 18; idx += 32) {
    int rr = idx / 18, cc = idx % 18;
    int gr = r0 + rr - 1, gc = c0 + cc - 1;
    float xv = 0.f;
    bool inb = (gr >= 0 && gr < Hs && gc >= 0 && gc < Ws);
    if (inb) {
      const float* fp = flow + (((size_t)bb * Hs + gr) * Ws + gc) * 2;
      Taps t = mkTaps((float)gc + fp[0], (float)gr + fp[1], Ws, Hs);
      xv = gatherC(img, bb, c, 3, Hs, Ws, t);
#if USE_ASYNC_LDS
      __builtin_amdgcn_global_load_async_to_lds_b32(
          (int*)(ref + (((size_t)bb * Hs + gr) * Ws + gc) * 3 + c),
          (int*)&Y[rr][cc], 0, 0);
#else
      Y[rr][cc] = ref[(((size_t)bb * Hs + gr) * Ws + gc) * 3 + c];
#endif
    } else {
      Y[rr][cc] = 0.f;
    }
    X[rr][cc] = xv;
  }
#if USE_ASYNC_LDS
  __builtin_amdgcn_s_wait_asynccnt(0);
#endif
  __syncthreads();

  // Build B (32x16 h-sum matrices) in WMMA B layout: lane -> column, 16 K rows
  v16h Bx{}, By{}, Bxx{}, Byy{}, Bxy{}, A{};
  const int j  = lane & 15;
  const int kb = (lane < 16) ? 0 : 16;
  for (int t = 0; t < 16; ++t) {
    int k = kb + t;
    float x0 = X[k][j], x1 = X[k][j+1], x2 = X[k][j+2];
    float y0 = Y[k][j], y1 = Y[k][j+1], y2 = Y[k][j+2];
    Bx[t]  = (_Float16)(x0 + x1 + x2);
    By[t]  = (_Float16)(y0 + y1 + y2);
    Bxx[t] = (_Float16)(x0*x0 + x1*x1 + x2*x2);
    Byy[t] = (_Float16)(y0*y0 + y1*y1 + y2*y2);
    Bxy[t] = (_Float16)(x0*y0 + x1*y1 + x2*y2);
  }
  // Banded A (16x32): A[m][k] = 1 for k in {m, m+1, m+2}; A uses k = 16*(lane/16) + t, same order as B
  const int m = lane & 15;
  for (int t = 0; t < 16; ++t) {
    int K = t + ((lane < 16) ? 0 : 16);
    A[t] = (_Float16)((K >= m && K <= m + 2) ? 1.0f : 0.0f);
  }
  v8f z = {0.f, 0.f, 0.f, 0.f, 0.f, 0.f, 0.f, 0.f};
  v8f Dx  = __builtin_amdgcn_wmma_f32_16x16x32_f16(false, A, false, Bx,  (short)0, z, false, false);
  v8f Dy  = __builtin_amdgcn_wmma_f32_16x16x32_f16(false, A, false, By,  (short)0, z, false, false);
  v8f Dxx = __builtin_amdgcn_wmma_f32_16x16x32_f16(false, A, false, Bxx, (short)0, z, false, false);
  v8f Dyy = __builtin_amdgcn_wmma_f32_16x16x32_f16(false, A, false, Byy, (short)0, z, false, false);
  v8f Dxy = __builtin_amdgcn_wmma_f32_16x16x32_f16(false, A, false, Bxy, (short)0, z, false, false);

  const float C1 = 1e-4f, C2 = 9e-4f;
  float lsum = 0.f;
  const int n  = lane & 15;
  const int mb = (lane < 16) ? 0 : 8;
  for (int q = 0; q < 8; ++q) {
    int mr = mb + q;
    int gr = r0 + mr, gc = c0 + n;
    bool ok = (gr < Hs) && (gc < Ws);
    int rlo = max(gr - 1, 0), rhi = min(gr + 1, Hs - 1);
    int clo = max(gc - 1, 0), chi = min(gc + 1, Ws - 1);
    float inv = 1.f / (float)((rhi - rlo + 1) * (chi - clo + 1));
    float mx = Dx[q] * inv, my = Dy[q] * inv;
    float sx  = Dxx[q] * inv - mx * mx;
    float syv = Dyy[q] * inv - my * my;
    float sxy = Dxy[q] * inv - mx * my;
    float nn = (2.f * mx * my + C1) * (2.f * sxy + C2);
    float dd = (mx * mx + my * my + C1) * (sx + syv + C2);
    float ss = fminf(fmaxf((1.f - nn / dd) * 0.5f, 0.f), 1.f);
    float xpt = X[mr + 1][n + 1], ypt = Y[mr + 1][n + 1];
    float err = 0.85f * ss + 0.15f * fabsf(xpt - ypt);
    if (ok) {
      if (isS0) out[(((size_t)bb * Hs + gr) * Ws + gc) * OUT_C + chbase + c] = err;
      else      lsum += err;
    }
  }
  if (!isS0) {
    float w = waveSum(lsum);
    if (lane == 0) atomicAdd(&acc[accIdx], w);
  }
}

// flow consistency + noc masks (+ fwd_full_warp channels at scale 0)
__global__ void k_fc(const float* __restrict__ FF, const float* __restrict__ BF,
                     const float* __restrict__ srcA, float* __restrict__ out,
                     float* __restrict__ acc, int s, int Hs, int Ws,
                     int isS0, float twoS) {
  int idx0 = blockIdx.x * blockDim.x + threadIdx.x;
  int np = Hs * Ws;
  int idx = min(idx0, np - 1);
  bool valid = (idx0 < np);
  int bb = blockIdx.y;
  int gr = idx / Ws, gc = idx % Ws;
  size_t po = ((size_t)bb * Hs + gr) * Ws + gc;
  float fx = FF[po*2], fy = FF[po*2+1];
  float bx = BF[po*2], by = BF[po*2+1];
  Taps tf = mkTaps((float)gc + fx, (float)gr + fy, Ws, Hs);
  Taps tb = mkTaps((float)gc + bx, (float)gr + by, Ws, Hs);
  float b2f0 = gatherC(BF, bb, 0, 2, Hs, Ws, tf);
  float b2f1 = gatherC(BF, bb, 1, 2, Hs, Ws, tf);
  float f2b0 = gatherC(FF, bb, 0, 2, Hs, Ws, tb);
  float f2b1 = gatherC(FF, bb, 1, 2, Hs, Ws, tb);
  float fd0 = fabsf(b2f0 + fx), fd1 = fabsf(b2f1 + fy);
  float bd0 = fabsf(f2b0 + bx), bd1 = fabsf(f2b1 + by);
  float fb  = fmaxf(0.05f * l2n2(fx, fy) * twoS, 3.0f);
  float bbt = fmaxf(0.05f * l2n2(bx, by) * twoS, 3.0f);
  float nt = (l2n2(fd0, fd1) * twoS < fb)  ? 1.f : 0.f;
  float ns = (l2n2(bd0, bd1) * twoS < bbt) ? 1.f : 0.f;
  if (isS0) {
    if (valid) {
      size_t oo = po * OUT_C;
      for (int c = 0; c < 3; ++c)
        out[oo + 12 + c] = gatherC(srcA, bb, c, 3, Hs, Ws, tf);
      out[oo + 15] = fd0; out[oo + 16] = fd1;
      out[oo + 17] = bd0; out[oo + 18] = bd1;
      out[oo + 19] = nt;  out[oo + 20] = ns;
    }
  } else {
    float vF = valid ? (fd0 + fd1) * nt : 0.f;
    float vB = valid ? (bd0 + bd1) * ns : 0.f;
    float wF = waveSum(vF), wB = waveSum(vB);
    if ((threadIdx.x & 31) == 0) {
      atomicAdd(&acc[12 + s - 1], wF);
      atomicAdd(&acc[15 + s - 1], wB);
    }
  }
}

__global__ void k_aux(float* __restrict__ acc) {
  if (threadIdx.x != 0 || blockIdx.x != 0) return;
  float aux = 0.f;
  for (int s = 1; s < 4; ++s) {
    float Hs = (float)(H_FULL >> s), Ws = (float)(W_FULL >> s);
    float np3 = 8.f * Hs * Ws * 3.f;
    float np2 = 8.f * Hs * Ws * 2.f;
    for (int p = 0; p < 4; ++p) aux += acc[(s - 1) * 4 + p] / np3;
    aux += acc[12 + s - 1] / np2;
    aux += acc[15 + s - 1] / np2;
  }
  acc[20] = aux;
}

// vectorized broadcast-add: b128 loads/stores keep this pass bandwidth-bound
__global__ void k_add4(float4* __restrict__ out4, const float* __restrict__ acc,
                       int n4, float* __restrict__ out, int nTail, int tailBase) {
  int i = blockIdx.x * blockDim.x + threadIdx.x;
  float a = acc[20];
  if (i < n4) {
    float4 v = out4[i];
    v.x += a; v.y += a; v.z += a; v.w += a;
    out4[i] = v;
  }
  if (i == 0) {
    for (int r = 0; r < nTail; ++r) out[tailBase + r] += a;
  }
}

// -------------------- launch --------------------
extern "C" void kernel_launch(void* const* d_in, const int* in_sizes, int n_in,
                              void* d_out, int out_size, void* d_ws, size_t ws_size,
                              hipStream_t stream) {
  (void)in_sizes; (void)n_in; (void)ws_size;
  const float* tgt   = (const float*)d_in[0];
  const float* src   = (const float*)d_in[1];
  const float* intr  = (const float*)d_in[2];
  const float* disp  = (const float*)d_in[3];
  const float* poses = (const float*)d_in[4];
  const float* fres  = (const float*)d_in[5];
  const float* bres  = (const float*)d_in[6];
  float* out = (float*)d_out;
  float* wsf = (float*)d_ws;

  float* ACC  = wsf + ACC_OFF;
  float* PF   = wsf + PF_OFF;
  float* PB   = wsf + PB_OFF;
  float* KI   = wsf + KI_OFF;
  float* TGTA = wsf + TGT_OFF;
  float* SRCA = wsf + SRC_OFF;
  float* FRp  = wsf + FR_OFF;
  float* BRp  = wsf + BR_OFF;
  float* FFp  = wsf + FF_OFF;
  float* BFp  = wsf + BF_OFF;

  k_zero<<<1, 64, 0, stream>>>(ACC);
  k_pose<<<1, 32, 0, stream>>>(intr, poses, PF, PB, KI);

  for (int s = 0; s < 4; ++s) {
    int Hs = H_FULL >> s, Ws = W_FULL >> s;
    int np = Hs * Ws;
    dim3 gp((np + 255) / 256, 8);
    k_prep<<<gp, 256, 0, stream>>>(tgt, src, disp, fres, bres, PF, PB, KI,
                                   TGTA, SRCA, FRp, BRp, FFp, BFp, s, Hs, Ws);

    int tilesX = (Ws + 15) / 16, tilesY = (Hs + 15) / 16;
    dim3 gs(tilesX * tilesY, 24);
    int isS0 = (s == 0) ? 1 : 0;
    int ab = (s >= 1) ? (s - 1) * 4 : 0;
    // pair 0: fwd_rigid_warp(src) vs tgt  -> ch 0..2
    k_ssim<<<gs, 32, 0, stream>>>(SRCA, TGTA, FRp, out, ACC, ab + 0, Hs, Ws, 0, isS0, tilesX);
    // pair 1: bwd_rigid_warp(tgt) vs src  -> ch 3..5
    k_ssim<<<gs, 32, 0, stream>>>(TGTA, SRCA, BRp, out, ACC, ab + 1, Hs, Ws, 3, isS0, tilesX);
    // pair 2: fwd_full_warp(src) vs tgt   -> ch 6..8
    k_ssim<<<gs, 32, 0, stream>>>(SRCA, TGTA, FFp, out, ACC, ab + 2, Hs, Ws, 6, isS0, tilesX);
    // pair 3: bwd_full_warp(tgt) vs src   -> ch 9..11
    k_ssim<<<gs, 32, 0, stream>>>(TGTA, SRCA, BFp, out, ACC, ab + 3, Hs, Ws, 9, isS0, tilesX);

    k_fc<<<gp, 256, 0, stream>>>(FFp, BFp, SRCA, out, ACC, s, Hs, Ws, isS0,
                                 (float)(1 << s));
  }

  k_aux<<<1, 1, 0, stream>>>(ACC);
  int n4 = out_size / 4;
  int tailBase = n4 * 4;
  int nTail = out_size - tailBase;
  k_add4<<<(n4 + 255) / 256, 256, 0, stream>>>((float4*)out, ACC, n4, out, nTail, tailBase);
}
